// GCN_22651657519777
// MI455X (gfx1250) — compile-verified, matrix operand order checked by reading, not run on hardware
//
#include <hip/hip_runtime.h>

#define NN 50000            // nodes
#define NE 1600000          // edges

typedef __attribute__((ext_vector_type(16))) __bf16 v16bf;
typedef __attribute__((ext_vector_type(8)))  __bf16 v8bf;
typedef __attribute__((ext_vector_type(8)))  float  v8f;

// ---------------- norm prep ----------------
__global__ void fill_one(float* p, int n) {
    int i = blockIdx.x * blockDim.x + threadIdx.x;
    if (i < n) p[i] = 1.0f;                       // self-loop contributes 1 to degree
}

__global__ void deg_count(const long long* __restrict__ dst, float* __restrict__ deg, int e) {
    int i = blockIdx.x * blockDim.x + threadIdx.x;
    if (i < e) atomicAdd(&deg[(int)dst[i]], 1.0f);
}

__global__ void rsqrt_ip(float* p, int n) {
    int i = blockIdx.x * blockDim.x + threadIdx.x;
    if (i < n) p[i] = rsqrtf(p[i]);               // deg >= 1 always
}

// --------------- weight prep: W[K][N] f32 -> Wt[N][K] bf16 ---------------
template <int K, int N>
__global__ void wprep(const float* __restrict__ W, __bf16* __restrict__ Wt) {
    int i = blockIdx.x * blockDim.x + threadIdx.x;
    if (i >= K * N) return;
    int k = i / N, n = i % N;
    Wt[n * K + k] = (__bf16)W[k * N + n];
}

// --------------- WMMA GEMM: H = relu?(A) @ W, O = H*dis^2 + bias ---------------
// A : M x K   (f32 row major, optionally ReLU'd, converted to bf16 in-register)
// Bt: NC x K  (bf16, = W transposed)
// One wave per (16 x 16*NT) output strip: A fragment loaded/converted once per
// k-step and fed to NT independent WMMAs (separate accumulators).
template <int K, int NC, int NT, bool RELU>
__global__ void gemm_wmma(const float* __restrict__ A,
                          const __bf16* __restrict__ Bt,
                          const float* __restrict__ dis,
                          const float* __restrict__ bias,
                          float* __restrict__ H,
                          float* __restrict__ O,
                          int M) {
    const int lane  = threadIdx.x;                           // 0..31
    const int mtile = blockIdx.x * blockDim.y + threadIdx.y;
    const int nt0   = blockIdx.y * NT;                       // first N-tile of this wave
    if (mtile * 16 >= M) return;                             // wave-uniform exit (EXEC all-1s)

    const int r     = mtile * 16 + (lane & 15);              // A row for this lane
    const int nb    = nt0 * 16 + (lane & 15);                // B column (tile 0) for this lane
    const int khalf = (lane >> 4) * 8;                       // lanes 16-31 take K-offset 8 / 24

    v8f acc[NT] = {};

#pragma unroll 2
    for (int k0 = 0; k0 < K; k0 += 32) {
        const int kb = k0 + khalf;
        // A fragment: K in [kb,kb+8) and [kb+16,kb+24), contiguous 32B runs
        const float* ap = A + (size_t)r * K + kb;
        v8f fa = *(const v8f*)(ap);
        v8f fb = *(const v8f*)(ap + 16);
        if (RELU) {
#pragma unroll
            for (int i = 0; i < 8; ++i) {
                fa[i] = fmaxf(fa[i], 0.0f);
                fb[i] = fmaxf(fb[i], 0.0f);
            }
        }
        v8bf a0 = __builtin_convertvector(fa, v8bf);
        v8bf a1 = __builtin_convertvector(fb, v8bf);
        v16bf av = __builtin_shufflevector(a0, a1, 0,1,2,3,4,5,6,7,8,9,10,11,12,13,14,15);

#pragma unroll
        for (int t = 0; t < NT; ++t) {
            const __bf16* bp = Bt + (size_t)(nb + 16 * t) * K + kb;
            v8bf b0 = *(const v8bf*)(bp);
            v8bf b1 = *(const v8bf*)(bp + 16);
            v16bf bv = __builtin_shufflevector(b0, b1, 0,1,2,3,4,5,6,7,8,9,10,11,12,13,14,15);
            acc[t] = __builtin_amdgcn_wmma_f32_16x16x32_bf16(false, av, false, bv,
                                                             (short)0, acc[t], false, false);
        }
    }

    // C/D layout: VGPR i -> row mtile*16 + i + 8*(lane>=16), col = ntile*16 + (lane&15)
    const int rbase = mtile * 16 + (lane >> 4) * 8;
#pragma unroll
    for (int t = 0; t < NT; ++t) {
        const int col  = nb + 16 * t;
        const float bb = bias[col];
#pragma unroll
        for (int i = 0; i < 8; ++i) {
            int row   = rbase + i;
            float h   = acc[t][i];
            size_t ix = (size_t)row * NC + col;
            H[ix] = h;                                // raw transform (gather source)
            float di = dis[row];
            O[ix] = fmaf(h * di, di, bb);             // self-loop term + bias pre-folded
        }
    }
}

// --------------- edge aggregation: O[dst] += H[src] * dis[src]*dis[dst] ---------------
template <int F>
__global__ void edge_agg(const long long* __restrict__ src,
                         const long long* __restrict__ dst,
                         const float* __restrict__ dis,
                         const float* __restrict__ H,
                         float* __restrict__ O) {
    constexpr int C = F / 4;                          // float4 chunks per edge
    long long tid = (long long)blockIdx.x * blockDim.x + threadIdx.x;
    if (tid >= (long long)NE * C) return;
    int c       = (int)(tid % C);
    long long e = tid / C;
    int s = (int)src[e], d = (int)dst[e];
    float w = dis[s] * dis[d];
    float4 h = *(const float4*)(H + (size_t)s * F + c * 4);
    float* o = O + (size_t)d * F + c * 4;
    atomicAdd(o + 0, h.x * w);
    atomicAdd(o + 1, h.y * w);
    atomicAdd(o + 2, h.z * w);
    atomicAdd(o + 3, h.w * w);
}

// ---------------- driver ----------------
extern "C" void kernel_launch(void* const* d_in, const int* in_sizes, int n_in,
                              void* d_out, int out_size, void* d_ws, size_t ws_size,
                              hipStream_t stream) {
    const float*     x   = (const float*)d_in[0];
    const long long* ei  = (const long long*)d_in[1];   // [2, NE] int64
    const float*     W1  = (const float*)d_in[2];
    const float*     b1  = (const float*)d_in[3];
    const float*     W2  = (const float*)d_in[4];
    const float*     b2  = (const float*)d_in[5];
    const float*     W3  = (const float*)d_in[6];
    const float*     b3  = (const float*)d_in[7];
    float* out = (float*)d_out;

    const long long* src = ei;
    const long long* dst = ei + NE;

    // workspace layout (bytes, 256-aligned)
    char* ws = (char*)d_ws;
    float*  dis = (float*)(ws);                                   // 200,000 B
    __bf16* w1t = (__bf16*)(ws + 200192);                         // 256*512*2
    __bf16* w2t = (__bf16*)(ws + 200192 + 262144);                // 32*256*2
    __bf16* w3t = (__bf16*)(ws + 200192 + 262144 + 16384);        // 16*32*2
    char*   big = ws + 479744;
    float* H1 = (float*)(big);                                    // 50000*256*4 = 51.2 MB
    float* O1 = (float*)(big + 51200000);                         // 51.2 MB
    // layer 2/3 buffers alias the H1 region (H1 dead before these are written)
    float* H2 = (float*)(big);                                    // 6.4 MB
    float* O2 = (float*)(big + 6400000);                          // 6.4 MB
    float* H3 = (float*)(big + 12800000);                         // 3.2 MB

    const int B = 256;
    dim3 wave(32, 4);
    const int mgrid = (NN / 16 + 3) / 4;                          // 3125 M-tiles / 4 waves

    // 1) symmetric norm: deg = 1 + count(dst);  dis = rsqrt(deg)
    fill_one<<<(NN + B - 1) / B, B, 0, stream>>>(dis, NN);
    deg_count<<<(NE + B - 1) / B, B, 0, stream>>>(dst, dis, NE);
    rsqrt_ip<<<(NN + B - 1) / B, B, 0, stream>>>(dis, NN);

    // 2) weight transpose+cast (tiny)
    wprep<512, 256><<<(512 * 256 + B - 1) / B, B, 0, stream>>>(W1, w1t);
    wprep<256, 32><<<(256 * 32 + B - 1) / B, B, 0, stream>>>(W2, w2t);
    wprep<32, 16><<<(32 * 16 + B - 1) / B, B, 0, stream>>>(W3, w3t);

    // 3) layer 1: GEMM (x) -> aggregate   [ReLU folded into layer-2 A load]
    gemm_wmma<512, 256, 4, false><<<dim3(mgrid, 4), wave, 0, stream>>>(x, w1t, dis, b1, H1, O1, NN);
    edge_agg<256><<<(unsigned)(((long long)NE * 64 + B - 1) / B), B, 0, stream>>>(src, dst, dis, H1, O1);

    // 4) layer 2: GEMM (relu(O1)) -> aggregate
    gemm_wmma<256, 32, 2, true><<<dim3(mgrid, 1), wave, 0, stream>>>(O1, w2t, dis, b2, H2, O2, NN);
    edge_agg<32><<<(unsigned)(((long long)NE * 8 + B - 1) / B), B, 0, stream>>>(src, dst, dis, H2, O2);

    // 5) layer 3: GEMM (relu(O2)) -> aggregate -> d_out (no ReLU on output)
    gemm_wmma<32, 16, 1, true><<<dim3(mgrid, 1), wave, 0, stream>>>(O2, w3t, dis, b3, H3, out, NN);
    edge_agg<16><<<(unsigned)(((long long)NE * 4 + B - 1) / B), B, 0, stream>>>(src, dst, dis, H3, out);
}
